// MambaEEG_55155970015239
// MI455X (gfx1250) — compile-verified, matrix operand order checked by reading, not run on hardware
//
#include <hip/hip_runtime.h>
#include <hip/hip_bf16.h>
#include <math.h>

#ifndef __has_builtin
#define __has_builtin(x) 0
#endif

// ---- problem constants (match reference) ----
#define D_MODEL 24
#define D_INNER 600
#define D_STATE 16
#define D_CONV  4
#define DT_RANK 2
#define LSEQ    250
#define C_IN    22
#define NCLS    4
#define BATCHN  256
#define MROWS   (BATCHN * LSEQ)     // 64000 token rows, multiple of 16

// padded GEMM dims
#define KPAD_DM  32                 // d_model 24 -> 32
#define NPAD_XZ  1200               // 2*d_inner, already /16
#define KPAD_DI  608                // d_inner 600 -> 608 (19 * 32)
#define NPAD_PRJ 48                 // dt_rank + 2*d_state = 34 -> 48
#define NPAD_OUT 32                 // d_model 24 -> 32

typedef __attribute__((ext_vector_type(16))) __bf16 v16bf;
typedef __attribute__((ext_vector_type(8)))  float  v8f;
typedef unsigned int v4u __attribute__((ext_vector_type(4)));
typedef int          v8i __attribute__((ext_vector_type(8)));
typedef int          v4i __attribute__((ext_vector_type(4)));

// Use the Tensor Data Mover when the builtin is available (device pass only).
#if defined(__HIP_DEVICE_COMPILE__) && __has_builtin(__builtin_amdgcn_tensor_load_to_lds) && __has_builtin(__builtin_amdgcn_s_wait_tensorcnt)
#define USE_TDM 1
#else
#define USE_TDM 0
#endif

__device__ __forceinline__ float siluf(float x) { return x / (1.f + __expf(-x)); }

// ---------------------------------------------------------------------------
// Flat WMMA GEMM (one 16x16 tile per wave): C[M,Npad] = A[M,Kpad] * B,
// with B supplied transposed: Bt[n][k] row-major [Npad][Kpad], so both
// operand gathers share the CDNA5 16-bit A-matrix lane/K mapping.
// Used for the K=32 in-projection where LDS staging buys nothing.
// ---------------------------------------------------------------------------
__global__ void gemm_bf16_wmma(const __bf16* __restrict__ A,
                               const __bf16* __restrict__ Bt,
                               float* __restrict__ C,
                               int Mrows, int Npad, int Kpad)
{
    const int nTilesN = Npad >> 4;
    const int nTiles  = (Mrows >> 4) * nTilesN;
    const int tile    = blockIdx.x * 8 + (threadIdx.x >> 5);   // wave32
    if (tile >= nTiles) return;                                // wave-uniform exit

    const int m0   = (tile / nTilesN) << 4;
    const int n0   = (tile % nTilesN) << 4;
    const int lane = threadIdx.x & 31;
    const int half = lane >> 4;       // lane group 0-15 vs 16-31
    const int r    = lane & 15;

    const __bf16* arow = A  + (size_t)(m0 + r) * Kpad;   // row m0+r of A
    const __bf16* brow = Bt + (size_t)(n0 + r) * Kpad;   // col n0+r of B

    v8f acc = {};
    for (int kb = 0; kb < Kpad; kb += 32) {
        v16bf va, vb;
#pragma unroll
        for (int j = 0; j < 16; ++j) {
            // CDNA5 16-bit A layout: lanes 0-15 -> K in {0..7, 16..23},
            // lanes 16-31 -> K in {8..15, 24..31}, two values per VGPR.
            int k = kb + ((j < 8) ? (8 * half + j) : (16 + 8 * half + (j - 8)));
            va[j] = arow[k];
            vb[j] = brow[k];
        }
        acc = __builtin_amdgcn_wmma_f32_16x16x32_bf16(
            /*neg_a=*/false, va, /*neg_b=*/false, vb,
            /*c_mod=*/(short)0, acc, /*reuse_a=*/false, /*reuse_b=*/false);
    }

    // D layout: VGPR v holds M = v + 8*half, N = lane&15
    const int col = n0 + r;
#pragma unroll
    for (int v = 0; v < 8; ++v) {
        int row = m0 + v + 8 * half;
        C[(size_t)row * Npad + col] = acc[v];
    }
}

// ---------------------------------------------------------------------------
// LDS-staged WMMA GEMM for the K=608 projections. One block = one 16-row
// M tile; the 16 x KPAD_DI bf16 A tile (19456 B) is staged into LDS once —
// by the Tensor Data Mover when available — then each wave computes one
// 16-wide N tile (blockDim.x = 32 * nTilesN).
// ---------------------------------------------------------------------------
__global__ void __launch_bounds__(128)
gemm_bf16_wmma_staged(const __bf16* __restrict__ A,
                      const __bf16* __restrict__ Bt,
                      float* __restrict__ C,
                      int Npad)
{
    __shared__ __bf16 sA[16 * KPAD_DI];
    const int m0 = blockIdx.x << 4;

#if USE_TDM
    if (threadIdx.x < 32) {            // wave 0 issues the TDM descriptor
        typedef __attribute__((address_space(3))) void lds_void_t;
        unsigned lds_base = (unsigned)(size_t)(lds_void_t*)&sA[0];
        unsigned long long ga =
            (unsigned long long)(size_t)(A + (size_t)m0 * KPAD_DI);
        const unsigned dim0 = (KPAD_DI * 2u) / 4u;   // 304 dwords per row

        v4u g0; v8i g1;
        v4i z4 = {0, 0, 0, 0};
#if __clang_major__ >= 23
        v8i z8 = {0, 0, 0, 0, 0, 0, 0, 0};
#endif
        // D# group 0: count=1 (valid, user mode), lds_addr, 57-bit global
        // address, type=2 ("image") in bits [127:126].
        g0.x = 1u;
        g0.y = lds_base;
        g0.z = (unsigned)(ga & 0xffffffffu);
        g0.w = (unsigned)((ga >> 32) & 0x1ffffffu) | (2u << 30);
        // D# group 1: data_size=2 (4B units); 2-D tile: 304 dwords x 16 rows,
        // tensor_dim0 = tile_dim0 = tensor_dim0_stride = 304, dim1 = 16.
        g1[0] = (int)(2u << 16);                        // data_size
        g1[1] = (int)((dim0 & 0xffffu) << 16);          // tensor_dim0[15:0]
        g1[2] = (int)((dim0 >> 16) | (16u << 16));      // dim0 hi | tensor_dim1 lo
        g1[3] = (int)(dim0 << 16);                      // dim1 hi(0) | tile_dim0
        g1[4] = 16;                                     // tile_dim1 | tile_dim2(0)
        g1[5] = (int)dim0;                              // tensor_dim0_stride[31:0]
        g1[6] = 0;
        g1[7] = 0;
#if __clang_major__ >= 23
        __builtin_amdgcn_tensor_load_to_lds(g0, g1, z4, z4, z8, 0);
#else
        __builtin_amdgcn_tensor_load_to_lds(g0, g1, z4, z4, 0);
#endif
        __builtin_amdgcn_s_wait_tensorcnt(0);           // TENSORcnt == 0
    }
    __syncthreads();
#else
    for (int i = threadIdx.x; i < 16 * KPAD_DI; i += blockDim.x)
        sA[i] = A[(size_t)m0 * KPAD_DI + i];
    __syncthreads();
#endif

    const int wv      = threadIdx.x >> 5;
    const int nTilesN = Npad >> 4;
    if (wv < nTilesN) {                       // wave-uniform
        const int n0   = wv << 4;
        const int lane = threadIdx.x & 31;
        const int half = lane >> 4;
        const int r    = lane & 15;
        const __bf16* brow = Bt + (size_t)(n0 + r) * KPAD_DI;
        const __bf16* arow = &sA[r * KPAD_DI];
        __builtin_prefetch(brow, 0, 1);       // global_prefetch_b8

        v8f acc = {};
        for (int kb = 0; kb < KPAD_DI; kb += 32) {
            v16bf va, vb;
#pragma unroll
            for (int j = 0; j < 16; ++j) {
                int k = kb + ((j < 8) ? (8 * half + j) : (16 + 8 * half + (j - 8)));
                va[j] = arow[k];              // ds_load from LDS
                vb[j] = brow[k];              // global load (L2-resident weights)
            }
            acc = __builtin_amdgcn_wmma_f32_16x16x32_bf16(
                false, va, false, vb, (short)0, acc, false, false);
        }
        const int col = n0 + r;
#pragma unroll
        for (int v = 0; v < 8; ++v)
            C[(size_t)(m0 + v + 8 * half) * Npad + col] = acc[v];
    }
}

// ---------------------------------------------------------------------------
// Embedding: h[b,l,dm] = sum_c x[b,c,l] * emb_w[dm,c] + emb_b[dm]
// Written directly as bf16, K-padded to 32 (pad lanes zeroed every call).
// ---------------------------------------------------------------------------
__global__ void embed_kernel(const float* __restrict__ x,
                             const float* __restrict__ ew,
                             const float* __restrict__ eb,
                             __bf16* __restrict__ hbf)
{
    int i = blockIdx.x * blockDim.x + threadIdx.x;
    if (i >= MROWS * KPAD_DM) return;
    int dm = i & (KPAD_DM - 1);
    int m  = i >> 5;
    float v = 0.f;
    if (dm < D_MODEL) {
        int b = m / LSEQ, l = m % LSEQ;
        float acc = eb[dm];
        const float* xb = x + (size_t)b * C_IN * LSEQ + l;
        const float* wr = ew + dm * C_IN;
#pragma unroll
        for (int c = 0; c < C_IN; ++c) acc += xb[(size_t)c * LSEQ] * wr[c];
        v = acc;
    }
    hbf[i] = (__bf16)v;
}

// Convert fp32 weight [N, K] row-major into padded bf16 Bt [Npad, Kpad]
__global__ void prep_bt(const float* __restrict__ W, __bf16* __restrict__ Bt,
                        int N, int Npad, int K, int Kpad)
{
    int i = blockIdx.x * blockDim.x + threadIdx.x;
    if (i >= Npad * Kpad) return;
    int k = i % Kpad, n = i / Kpad;
    float v = (n < N && k < K) ? W[(size_t)n * K + k] : 0.f;
    Bt[i] = (__bf16)v;
}

// ---------------------------------------------------------------------------
// Depthwise causal conv (width 4) + bias + SiLU. Reads xi = xz[:, 0:600].
// Emits fp32 (for scan) and bf16 K-padded (x-proj GEMM operand).
// ---------------------------------------------------------------------------
__global__ void conv_silu_kernel(const float* __restrict__ xz,
                                 const float* __restrict__ cw,
                                 const float* __restrict__ cb,
                                 float* __restrict__ xs,
                                 __bf16* __restrict__ xsb)
{
    int i = blockIdx.x * blockDim.x + threadIdx.x;
    if (i >= MROWS * KPAD_DI) return;
    int d = i % KPAD_DI;
    int m = i / KPAD_DI;
    if (d < D_INNER) {
        int b = m / LSEQ, l = m % LSEQ;
        float acc = cb[d];
#pragma unroll
        for (int j = 0; j < D_CONV; ++j) {
            int li = l - (D_CONV - 1) + j;
            if (li >= 0)
                acc += xz[(size_t)(b * LSEQ + li) * NPAD_XZ + d] * cw[d * D_CONV + j];
        }
        float s = siluf(acc);
        xs[(size_t)m * D_INNER + d] = s;
        xsb[i] = (__bf16)s;
    } else {
        xsb[i] = (__bf16)0.f;   // K-pad lanes
    }
}

// delta = softplus(dt @ W_dt^T + b_dt), dt = proj[:, 0:2]
__global__ void delta_kernel(const float* __restrict__ proj,
                             const float* __restrict__ wdt,
                             const float* __restrict__ bdt,
                             float* __restrict__ delta)
{
    int i = blockIdx.x * blockDim.x + threadIdx.x;
    if (i >= MROWS * D_INNER) return;
    int d = i % D_INNER, m = i / D_INNER;
    float v = proj[(size_t)m * NPAD_PRJ + 0] * wdt[d * DT_RANK + 0]
            + proj[(size_t)m * NPAD_PRJ + 1] * wdt[d * DT_RANK + 1]
            + bdt[d];
    delta[i] = (v > 20.f) ? v : log1pf(__expf(v));
}

// ---------------------------------------------------------------------------
// Selective scan, fused with D-skip and SiLU(z) gate, writing the bf16
// out-proj operand (K-padded) directly. One block per batch element,
// 608 threads: 600 channels + 8 pad-writers. State lives in registers.
// ---------------------------------------------------------------------------
__global__ void __launch_bounds__(KPAD_DI)
scan_gate_kernel(const float* __restrict__ delta,
                 const float* __restrict__ xs,
                 const float* __restrict__ proj,
                 const float* __restrict__ Alog,
                 const float* __restrict__ Dp,
                 const float* __restrict__ xz,
                 __bf16* __restrict__ ybf)
{
    int b = blockIdx.x;
    int d = threadIdx.x;
    __shared__ float sB[D_STATE], sC[D_STATE];

    float A[D_STATE], h[D_STATE];
    float Dd = 0.f;
    if (d < D_INNER) {
        Dd = Dp[d];
#pragma unroll
        for (int n = 0; n < D_STATE; ++n) {
            A[n] = -__expf(Alog[d * D_STATE + n]);
            h[n] = 0.f;
        }
    }

    for (int l = 0; l < LSEQ; ++l) {
        int m = b * LSEQ + l;
        __syncthreads();
        if (d < D_STATE)
            sB[d] = proj[(size_t)m * NPAD_PRJ + DT_RANK + d];
        else if (d < 2 * D_STATE)
            sC[d - D_STATE] = proj[(size_t)m * NPAD_PRJ + DT_RANK + D_STATE + (d - D_STATE)];
        __syncthreads();

        if (d < D_INNER) {
            float dl = delta[(size_t)m * D_INNER + d];
            float xv = xs[(size_t)m * D_INNER + d];
            float dx = dl * xv;
            float y = 0.f;
#pragma unroll
            for (int n = 0; n < D_STATE; ++n) {
                h[n] = __expf(dl * A[n]) * h[n] + dx * sB[n];
                y += h[n] * sC[n];
            }
            y += xv * Dd;                                  // D skip
            float z = xz[(size_t)m * NPAD_XZ + D_INNER + d];
            y *= siluf(z);                                 // gate
            ybf[(size_t)m * KPAD_DI + d] = (__bf16)y;
        } else {
            ybf[(size_t)m * KPAD_DI + d] = (__bf16)0.f;    // K-pad lanes
        }
    }
}

// ---------------------------------------------------------------------------
// LayerNorm over (L, d_model) per batch element. One block per b.
// Optionally emits bf16 (next mamba's A operand, K-padded buffer whose
// pad lanes are already zero) and/or fp32 (in-place ok).
// ---------------------------------------------------------------------------
__global__ void ln2d_kernel(const float* __restrict__ hin,
                            const float* __restrict__ w,
                            const float* __restrict__ bias,
                            __bf16* __restrict__ outBf,
                            float* __restrict__ outF)
{
    int b = blockIdx.x, t = threadIdx.x;
    __shared__ float s1[256], s2[256];
    const int NEL = LSEQ * D_MODEL;
    float su = 0.f, sq = 0.f;
    for (int i = t; i < NEL; i += 256) {
        int l = i / D_MODEL, dm = i % D_MODEL;
        float v = hin[(size_t)(b * LSEQ + l) * NPAD_OUT + dm];
        su += v; sq += v * v;
    }
    s1[t] = su; s2[t] = sq;
    __syncthreads();
    for (int o = 128; o > 0; o >>= 1) {
        if (t < o) { s1[t] += s1[t + o]; s2[t] += s2[t + o]; }
        __syncthreads();
    }
    float mu  = s1[0] / NEL;
    float var = s2[0] / NEL - mu * mu;
    float rs  = rsqrtf(var + 1e-5f);
    for (int i = t; i < NEL; i += 256) {
        int l = i / D_MODEL, dm = i % D_MODEL;
        size_t idx = (size_t)(b * LSEQ + l) * NPAD_OUT + dm;
        float v = (hin[idx] - mu) * rs * w[i] + bias[i];
        if (outBf) outBf[(size_t)(b * LSEQ + l) * KPAD_DM + dm] = (__bf16)v;
        if (outF)  outF[idx] = v;
    }
}

// mean over L then FC head: out[b,c] = fc_b[c] + sum_dm mean_l(h[b,l,dm]) * fc_w[c,dm]
__global__ void head_kernel(const float* __restrict__ hin,
                            const float* __restrict__ fcw,
                            const float* __restrict__ fcb,
                            float* __restrict__ out)
{
    int b = blockIdx.x, t = threadIdx.x;
    __shared__ float mv[D_MODEL];
    if (t < D_MODEL) {
        float acc = 0.f;
        for (int l = 0; l < LSEQ; ++l)
            acc += hin[(size_t)(b * LSEQ + l) * NPAD_OUT + t];
        mv[t] = acc / (float)LSEQ;
    }
    __syncthreads();
    if (t < NCLS) {
        float acc = fcb[t];
#pragma unroll
        for (int dm = 0; dm < D_MODEL; ++dm) acc += mv[dm] * fcw[t * D_MODEL + dm];
        out[b * NCLS + t] = acc;
    }
}

// ---------------------------------------------------------------------------
// Host-side orchestration
// ---------------------------------------------------------------------------
static inline int cdiv(long long a, long long b) { return (int)((a + b - 1) / b); }

struct Ws {
    float  *xz, *xs, *delta, *proj, *hout;
    __bf16 *hbf, *xsb, *ybf, *wbtIn, *wbtX, *wbtOut;
};

static void run_mamba(const float* const* P, const Ws& w, hipStream_t stream)
{
    const float* inproj  = P[0];
    const float* convw   = P[1];
    const float* convb   = P[2];
    const float* xproj   = P[3];
    const float* dtw     = P[4];
    const float* dtb     = P[5];
    const float* Alog    = P[6];
    const float* Dp      = P[7];
    const float* outproj = P[8];

    // weights -> padded bf16 Bt
    prep_bt<<<cdiv((long long)NPAD_XZ * KPAD_DM, 256), 256, 0, stream>>>(
        inproj, w.wbtIn, 2 * D_INNER, NPAD_XZ, D_MODEL, KPAD_DM);
    prep_bt<<<cdiv((long long)NPAD_PRJ * KPAD_DI, 256), 256, 0, stream>>>(
        xproj, w.wbtX, DT_RANK + 2 * D_STATE, NPAD_PRJ, D_INNER, KPAD_DI);
    prep_bt<<<cdiv((long long)NPAD_OUT * KPAD_DI, 256), 256, 0, stream>>>(
        outproj, w.wbtOut, D_MODEL, NPAD_OUT, D_INNER, KPAD_DI);

    // in-proj GEMM: [M,32] x [32,1200] -> xz [M,1200]  (flat WMMA)
    {
        int tiles = (MROWS / 16) * (NPAD_XZ / 16);
        gemm_bf16_wmma<<<cdiv(tiles, 8), 256, 0, stream>>>(
            w.hbf, w.wbtIn, w.xz, MROWS, NPAD_XZ, KPAD_DM);
    }
    // causal conv + SiLU
    conv_silu_kernel<<<cdiv((long long)MROWS * KPAD_DI, 256), 256, 0, stream>>>(
        w.xz, convw, convb, w.xs, w.xsb);
    // x-proj GEMM: [M,608] x [608,48] -> proj [M,48]  (TDM/LDS staged)
    gemm_bf16_wmma_staged<<<MROWS / 16, 32 * (NPAD_PRJ / 16), 0, stream>>>(
        w.xsb, w.wbtX, w.proj, NPAD_PRJ);
    // delta
    delta_kernel<<<cdiv((long long)MROWS * D_INNER, 256), 256, 0, stream>>>(
        w.proj, dtw, dtb, w.delta);
    // selective scan + gate -> ybf (bf16)
    scan_gate_kernel<<<BATCHN, KPAD_DI, 0, stream>>>(
        w.delta, w.xs, w.proj, Alog, Dp, w.xz, w.ybf);
    // out-proj GEMM: [M,608] x [608,32] -> hout [M,32]  (TDM/LDS staged)
    gemm_bf16_wmma_staged<<<MROWS / 16, 32 * (NPAD_OUT / 16), 0, stream>>>(
        w.ybf, w.wbtOut, w.hout, NPAD_OUT);
}

extern "C" void kernel_launch(void* const* d_in, const int* in_sizes, int n_in,
                              void* d_out, int out_size, void* d_ws, size_t ws_size,
                              hipStream_t stream)
{
    const float* x     = (const float*)d_in[0];
    const float* emb_w = (const float*)d_in[1];
    const float* emb_b = (const float*)d_in[2];
    const float* m1[9];
    const float* m2[9];
    for (int i = 0; i < 9; ++i) m1[i] = (const float*)d_in[3 + i];
    for (int i = 0; i < 9; ++i) m2[i] = (const float*)d_in[12 + i];
    const float* ln1_w = (const float*)d_in[21];
    const float* ln1_b = (const float*)d_in[22];
    const float* ln2_w = (const float*)d_in[23];
    const float* ln2_b = (const float*)d_in[24];
    const float* fc_w  = (const float*)d_in[25];
    const float* fc_b  = (const float*)d_in[26];
    float* out = (float*)d_out;

    // carve workspace
    char* p = (char*)d_ws;
    auto carve = [&p](size_t bytes) -> void* {
        void* r = (void*)p;
        p += (bytes + 255) & ~(size_t)255;
        return r;
    };
    Ws w;
    w.xz     = (float*) carve((size_t)MROWS * NPAD_XZ  * sizeof(float));
    w.xs     = (float*) carve((size_t)MROWS * D_INNER  * sizeof(float));
    w.delta  = (float*) carve((size_t)MROWS * D_INNER  * sizeof(float));
    w.proj   = (float*) carve((size_t)MROWS * NPAD_PRJ * sizeof(float));
    w.hout   = (float*) carve((size_t)MROWS * NPAD_OUT * sizeof(float));
    w.hbf    = (__bf16*)carve((size_t)MROWS * KPAD_DM  * sizeof(__bf16));
    w.xsb    = (__bf16*)carve((size_t)MROWS * KPAD_DI  * sizeof(__bf16));
    w.ybf    = (__bf16*)carve((size_t)MROWS * KPAD_DI  * sizeof(__bf16));
    w.wbtIn  = (__bf16*)carve((size_t)NPAD_XZ  * KPAD_DM * sizeof(__bf16));
    w.wbtX   = (__bf16*)carve((size_t)NPAD_PRJ * KPAD_DI * sizeof(__bf16));
    w.wbtOut = (__bf16*)carve((size_t)NPAD_OUT * KPAD_DI * sizeof(__bf16));
    (void)ws_size; (void)in_sizes; (void)n_in; (void)out_size;

    // embedding -> bf16 padded A operand (also zeroes K-pad lanes every call)
    embed_kernel<<<cdiv((long long)MROWS * KPAD_DM, 256), 256, 0, stream>>>(
        x, emb_w, emb_b, w.hbf);

    // mamba block 1
    run_mamba(m1, w, stream);
    // LN1 -> bf16 input for block 2 (pad lanes of hbf still zero from embed)
    ln2d_kernel<<<BATCHN, 256, 0, stream>>>(w.hout, ln1_w, ln1_b, w.hbf, nullptr);

    // mamba block 2
    run_mamba(m2, w, stream);
    // LN2 in-place fp32
    ln2d_kernel<<<BATCHN, 256, 0, stream>>>(w.hout, ln2_w, ln2_b, nullptr, w.hout);

    // mean over L + FC head
    head_kernel<<<BATCHN, 32, 0, stream>>>(w.hout, fc_w, fc_b, out);
}
